// MultiHeadAttention_Enc_781684048473
// MI455X (gfx1250) — matrix operand order repaired and checked by reading, not compiled, not run on hardware
//
#include <hip/hip_runtime.h>
#include <hip/hip_bf16.h>
#include <stdint.h>

#define EMB   1024
#define HEADS 16
#define HDIM  64
#define SEQ   2048
#define BATCH 4
#define NROWS (BATCH * SEQ)   // 8192

typedef __attribute__((ext_vector_type(16))) __bf16 v16bf;
typedef __attribute__((ext_vector_type(8)))  __bf16 v8bf;
typedef __attribute__((ext_vector_type(8)))  float  v8f;

// ---- WMMA operand loaders (wave32; this code's contiguous k order, used alike for A and B) ----
// A operand (16-bit, 16x32): lane holds row (lane%16); 16 contiguous K values
// starting at K = (lane/16)*16.
__device__ __forceinline__ v16bf load_a_bf16(const __bf16* rowp, int half) {
    return *(const v16bf*)(rowp + half * 16);
}
// B operand (16-bit, 32x16): lane holds column (lane%16); 16 contiguous K values
// starting at K = (lane/16)*16.  rowp points at the source row == B column.
__device__ __forceinline__ v16bf load_b_bf16(const __bf16* rowp, int half) {
    return *(const v16bf*)(rowp + half * 16);
}
__device__ __forceinline__ v8f wmma_bf16(v16bf a, v16bf b, v8f c) {
    return __builtin_amdgcn_wmma_f32_16x16x32_bf16(false, a, false, b, (short)0, c,
                                                   false, false);
}

// ---------------------------------------------------------------------------
// Kernel 0: one-time f32 -> bf16 conversion (native fptrunc, RNE).
// ---------------------------------------------------------------------------
__global__ __launch_bounds__(256)
void cvt_f32_to_bf16(const float* __restrict__ src, __bf16* __restrict__ dst) {
    size_t i = ((size_t)blockIdx.x * 256 + threadIdx.x) * 8;
    float4 f0 = *(const float4*)(src + i);
    float4 f1 = *(const float4*)(src + i + 4);
    v8bf o;
    o[0] = (__bf16)f0.x; o[1] = (__bf16)f0.y; o[2] = (__bf16)f0.z; o[3] = (__bf16)f0.w;
    o[4] = (__bf16)f1.x; o[5] = (__bf16)f1.y; o[6] = (__bf16)f1.z; o[7] = (__bf16)f1.w;
    *(v8bf*)(dst + i) = o;
}

// ---------------------------------------------------------------------------
// Kernel 1: projection  dst = cvt_bf16( X[8192,1024] @ W[1024,1024]^T + bias )
// vmode==0: dst layout [b, h, n, d]   (Q, K)
// vmode==1: dst layout [b, h, d, n]   (V transposed, for P@V B-operand loads)
// One wave computes a 32x64 tile (8 WMMA accumulators); block = 4 waves = 64x128.
// ---------------------------------------------------------------------------
__global__ __launch_bounds__(128)
void qkv_proj_kernel(const __bf16* __restrict__ X, const __bf16* __restrict__ W,
                     const float* __restrict__ bias, __bf16* __restrict__ dst,
                     int vmode) {
    const int lane = threadIdx.x & 31;
    const int wave = threadIdx.x >> 5;
    const int half = lane >> 4;
    const int lrow = lane & 15;
    const int rbase = blockIdx.x * 64 + (wave >> 1) * 32;    // row block (n)
    const int cbase = blockIdx.y * 128 + (wave & 1) * 64;    // col block (o)

    const __bf16* arow0 = X + (size_t)(rbase + lrow) * EMB;
    const __bf16* arow1 = X + (size_t)(rbase + 16 + lrow) * EMB;
    const __bf16* brow[4];
#pragma unroll
    for (int tj = 0; tj < 4; ++tj)
        brow[tj] = W + (size_t)(cbase + tj * 16 + lrow) * EMB;

    v8f acc[2][4] = {};
    for (int k = 0; k < EMB; k += 32) {
        v16bf a0 = load_a_bf16(arow0 + k, half);
        v16bf a1 = load_a_bf16(arow1 + k, half);
#pragma unroll
        for (int tj = 0; tj < 4; ++tj) {
            v16bf b = load_b_bf16(brow[tj] + k, half);
            acc[0][tj] = wmma_bf16(a0, b, acc[0][tj]);
            acc[1][tj] = wmma_bf16(a1, b, acc[1][tj]);
        }
    }
#pragma unroll
    for (int ti = 0; ti < 2; ++ti) {
#pragma unroll
        for (int tj = 0; tj < 4; ++tj) {
            int o = cbase + tj * 16 + lrow;        // C-layout column = lane%16
            float bv = bias[o];
            int h = o >> 6, d = o & 63;
#pragma unroll
            for (int v = 0; v < 8; ++v) {
                int n = rbase + ti * 16 + v + half * 8;  // C-layout row
                int b = n >> 11, q = n & 2047;
                float val = acc[ti][tj][v] + bv;
                size_t idx;
                if (vmode) idx = ((size_t)(b * HEADS + h) * HDIM + d) * SEQ + q;
                else       idx = ((size_t)(b * HEADS + h) * SEQ + q) * HDIM + d;
                dst[idx] = (__bf16)val;
            }
        }
    }
}

// ---------------------------------------------------------------------------
// Kernel 2: flash attention, one wave per 16-query tile, 64 keys per step.
//   S = Q K^T ; P = exp2((S - m) * scale*log2e) ; O += P V ; osum += P @ 1
// Fused denominator: all-ones B operand makes every column of osum equal the
// row sum, so no cross-lane sum reductions and no end-of-loop broadcast.
// P goes C-layout -> LDS (bf16) -> A-operand layout (same-wave DS ordering).
// ---------------------------------------------------------------------------
__global__ __launch_bounds__(128)
void flash_attn_kernel(const __bf16* __restrict__ Qh, const __bf16* __restrict__ Kh,
                       const __bf16* __restrict__ Vt, __bf16* __restrict__ attO) {
    __shared__ __align__(16) __bf16 Pbuf[4][16 * 64];
    const int lane = threadIdx.x & 31;
    const int wave = threadIdx.x >> 5;
    const int half = lane >> 4;
    const int lrow = lane & 15;
    const int bh = blockIdx.y;                        // b*16 + h
    const int q0 = (blockIdx.x * 4 + wave) * 16;
    const float c2 = 0.04508422002778011f;            // (1/sqrt(EMB)) * log2(e)

    const __bf16* Qbase = Qh + (size_t)bh * SEQ * HDIM;
    const __bf16* Kbase = Kh + (size_t)bh * SEQ * HDIM;
    const __bf16* Vbase = Vt + (size_t)bh * HDIM * SEQ;

    // Q tile as A operand, split over d: [0..31] and [32..63]
    const __bf16* qrow = Qbase + (size_t)(q0 + lrow) * HDIM;
    v16bf aq0 = load_a_bf16(qrow, half);
    v16bf aq1 = load_a_bf16(qrow + 32, half);

    // all-ones B operand (32x16 of 1.0) for the fused row-sum accumulator
    v16bf bones;
#pragma unroll
    for (int i = 0; i < 16; ++i) bones[i] = (__bf16)1.0f;

    v8f o[4] = {};
    v8f osum = {};
    float m[8];
#pragma unroll
    for (int v = 0; v < 8; ++v) m[v] = -1e30f;

    __bf16* pb = &Pbuf[wave][0];

    for (int kb = 0; kb < SEQ; kb += 64) {
        // S: 4 key tiles of 16, each reduced over d = 64 (2 WMMAs)
        v8f s[4];
#pragma unroll
        for (int j = 0; j < 4; ++j) {
            const __bf16* krow = Kbase + (size_t)(kb + j * 16 + lrow) * HDIM;
            v8f sj = {};
            sj = wmma_bf16(aq0, load_b_bf16(krow, half), sj);
            sj = wmma_bf16(aq1, load_b_bf16(krow + 32, half), sj);
            s[j] = sj;
        }

        float corr[8];
#pragma unroll
        for (int v = 0; v < 8; ++v) {
            // row max over 64 keys: 4-way in-lane, then across the 16-lane group
            float t = fmaxf(fmaxf(s[0][v], s[1][v]), fmaxf(s[2][v], s[3][v]));
            t = fmaxf(t, __shfl_xor(t, 1, 32));
            t = fmaxf(t, __shfl_xor(t, 2, 32));
            t = fmaxf(t, __shfl_xor(t, 4, 32));
            t = fmaxf(t, __shfl_xor(t, 8, 32));
            float mn = fmaxf(m[v], t);
            float mc = mn * c2;
            corr[v] = exp2f((m[v] - mn) * c2);
            m[v] = mn;
            // P in exp2 domain; C-layout -> LDS row-major P[16][64]
            int r = (v + half * 8) * 64 + lrow;
#pragma unroll
            for (int j = 0; j < 4; ++j)
                pb[r + j * 16] = (__bf16)exp2f(fmaf(s[j][v], c2, -mc));
        }
#pragma unroll
        for (int v = 0; v < 8; ++v) {
            o[0][v] *= corr[v]; o[1][v] *= corr[v];
            o[2][v] *= corr[v]; o[3][v] *= corr[v];
            osum[v] *= corr[v];
        }
        // same-wave DS ops are in-order; fence + wait before re-reading P
        asm volatile("s_wait_dscnt 0" ::: "memory");
        v16bf ap0 = load_a_bf16(pb + (size_t)lrow * 64, half);       // keys kb..+31
        v16bf ap1 = load_a_bf16(pb + (size_t)lrow * 64 + 32, half);  // keys kb+32..+63
        // O += P @ V : B column = d, from Vt rows (contiguous over keys)
#pragma unroll
        for (int t = 0; t < 4; ++t) {
            const __bf16* vrow = Vbase + (size_t)(t * 16 + lrow) * SEQ + kb;
            o[t] = wmma_bf16(ap0, load_b_bf16(vrow, half), o[t]);
            o[t] = wmma_bf16(ap1, load_b_bf16(vrow + 32, half), o[t]);
        }
        osum = wmma_bf16(ap0, bones, osum);
        osum = wmma_bf16(ap1, bones, osum);
    }

    const int b = bh >> 4, h = bh & 15;
#pragma unroll
    for (int v = 0; v < 8; ++v) {
        float inv = 1.0f / osum[v];   // every lane holds its rows' denominator
        int q = q0 + v + half * 8;
        size_t nrow = ((size_t)b * SEQ + q) * EMB + (size_t)h * HDIM;
        attO[nrow + 0  + lrow] = (__bf16)(o[0][v] * inv);
        attO[nrow + 16 + lrow] = (__bf16)(o[1][v] * inv);
        attO[nrow + 32 + lrow] = (__bf16)(o[2][v] * inv);
        attO[nrow + 48 + lrow] = (__bf16)(o[3][v] * inv);
    }
}

// ---------------------------------------------------------------------------
// Kernel 3: output projection  out(f32) = attO(bf16) @ Wp(bf16)^T + bp
// ---------------------------------------------------------------------------
__global__ __launch_bounds__(128)
void out_proj_kernel(const __bf16* __restrict__ A, const __bf16* __restrict__ W,
                     const float* __restrict__ bias, float* __restrict__ out) {
    const int lane = threadIdx.x & 31;
    const int wave = threadIdx.x >> 5;
    const int half = lane >> 4;
    const int lrow = lane & 15;
    const int rbase = blockIdx.x * 64 + (wave >> 1) * 32;
    const int cbase = blockIdx.y * 128 + (wave & 1) * 64;

    const __bf16* arow0 = A + (size_t)(rbase + lrow) * EMB;
    const __bf16* arow1 = A + (size_t)(rbase + 16 + lrow) * EMB;
    const __bf16* brow[4];
#pragma unroll
    for (int tj = 0; tj < 4; ++tj)
        brow[tj] = W + (size_t)(cbase + tj * 16 + lrow) * EMB;

    v8f acc[2][4] = {};
    for (int k = 0; k < EMB; k += 32) {
        v16bf a0 = load_a_bf16(arow0 + k, half);
        v16bf a1 = load_a_bf16(arow1 + k, half);
#pragma unroll
        for (int tj = 0; tj < 4; ++tj) {
            v16bf b = load_b_bf16(brow[tj] + k, half);
            acc[0][tj] = wmma_bf16(a0, b, acc[0][tj]);
            acc[1][tj] = wmma_bf16(a1, b, acc[1][tj]);
        }
    }
#pragma unroll
    for (int ti = 0; ti < 2; ++ti) {
#pragma unroll
        for (int tj = 0; tj < 4; ++tj) {
            int o = cbase + tj * 16 + lrow;
            float bv = bias[o];
#pragma unroll
            for (int v = 0; v < 8; ++v) {
                int n = rbase + ti * 16 + v + half * 8;
                out[(size_t)n * EMB + o] = acc[ti][tj][v] + bv;
            }
        }
    }
}

// ---------------------------------------------------------------------------
extern "C" void kernel_launch(void* const* d_in, const int* in_sizes, int n_in,
                              void* d_out, int out_size, void* d_ws, size_t ws_size,
                              hipStream_t stream) {
    (void)in_sizes; (void)n_in; (void)out_size; (void)ws_size;
    const float* X  = (const float*)d_in[0] + (size_t)BATCH * SEQ * EMB;  // X[1]
    const float* Wq = (const float*)d_in[1];
    const float* bq = (const float*)d_in[2];
    const float* Wk = (const float*)d_in[3];
    const float* bk = (const float*)d_in[4];
    const float* Wv = (const float*)d_in[5];
    const float* bv = (const float*)d_in[6];
    const float* Wp = (const float*)d_in[7];
    const float* bp = (const float*)d_in[8];

    __bf16* Xbf  = (__bf16*)d_ws;                       // [n, e]     16 MB
    __bf16* Wqb  = Xbf + (size_t)NROWS * EMB;           // [o, e]      2 MB
    __bf16* Wkb  = Wqb + (size_t)EMB * EMB;
    __bf16* Wvb  = Wkb + (size_t)EMB * EMB;
    __bf16* Wpb  = Wvb + (size_t)EMB * EMB;
    __bf16* Qh   = Wpb + (size_t)EMB * EMB;             // [b,h,n,d]  16 MB
    __bf16* Kh   = Qh + (size_t)NROWS * EMB;            // [b,h,n,d]  16 MB
    __bf16* Vt   = Kh + (size_t)NROWS * EMB;            // [b,h,d,n]  16 MB
    __bf16* attO = Vt + (size_t)NROWS * EMB;            // [n, e]     16 MB

    // one-time conversions (sizes are exact multiples of 8*256)
    cvt_f32_to_bf16<<<(NROWS * EMB) / (8 * 256), 256, 0, stream>>>(X, Xbf);
    cvt_f32_to_bf16<<<(EMB * EMB) / (8 * 256), 256, 0, stream>>>(Wq, Wqb);
    cvt_f32_to_bf16<<<(EMB * EMB) / (8 * 256), 256, 0, stream>>>(Wk, Wkb);
    cvt_f32_to_bf16<<<(EMB * EMB) / (8 * 256), 256, 0, stream>>>(Wv, Wvb);
    cvt_f32_to_bf16<<<(EMB * EMB) / (8 * 256), 256, 0, stream>>>(Wp, Wpb);

    dim3 gproj(NROWS / 64, EMB / 128);
    qkv_proj_kernel<<<gproj, 128, 0, stream>>>(Xbf, Wqb, bq, Qh, 0);
    qkv_proj_kernel<<<gproj, 128, 0, stream>>>(Xbf, Wkb, bk, Kh, 0);
    qkv_proj_kernel<<<gproj, 128, 0, stream>>>(Xbf, Wvb, bv, Vt, 1);

    flash_attn_kernel<<<dim3(SEQ / 64, BATCH * HEADS), 128, 0, stream>>>(Qh, Kh, Vt, attO);

    out_proj_kernel<<<gproj, 128, 0, stream>>>(attO, Wpb, bp, (float*)d_out);
}
